// NeuralDecisionTreeRegression_64278480552035
// MI455X (gfx1250) — matrix operand order repaired and checked by reading, not compiled, not run on hardware
//
#include <hip/hip_runtime.h>
#include <hip/hip_bf16.h>
#include <math.h>

typedef __attribute__((ext_vector_type(16))) _Float16 v16h;
typedef __attribute__((ext_vector_type(8)))  _Float16 v8h;
typedef __attribute__((ext_vector_type(4)))  _Float16 v4h;
typedef __attribute__((ext_vector_type(8)))  float    v8f;
typedef __attribute__((ext_vector_type(4)))  float    v4f;

#define IN_DIM 500
#define K1PAD  512
#define H1 256
#define H2 128
#define H3 64
#define BN_EPS 1e-3f

// ---- workspace byte offsets ----
#define WS_W1H   0          // 256*512 f16  = 262144 B
#define WS_W2H   262144     // 128*256 f16  =  65536 B
#define WS_W3H   327680     //  64*128 f16  =  16384 B
#define WS_WNH   344064     //  32*64  f16  =   4096 B
#define WS_A1    348160     // 256 f32
#define WS_C1    349184
#define WS_A2    350208     // 128 f32
#define WS_C2    350720
#define WS_A3    351232     //  64 f32
#define WS_C3    351488

// ---- dynamic LDS layout (bytes) ----
#define SM_A     0          // 128 x 264 halves (act1) = 67584 B ; reused for act3 + nodes
#define SM_NODES 20480      // 128 x 8 f32 = 4096 B (inside region A, past act3's 18432 B)
#define SM_B     67584      // 128 x 136 halves (xbuf / act2) = 34816 B
#define SM_P     102400     // params: 2049 f32 (padded)
#define SMEM_BYTES 110848

// ---- param f32 offsets inside SM_P ----
#define P_B1 0
#define P_A1 256
#define P_C1 512
#define P_B2 768
#define P_A2 896
#define P_C2 1024
#define P_B3 1152
#define P_A3 1216
#define P_C3 1280
#define P_THR 1344
#define P_TEMP 1376
#define P_LEAF 1408
#define P_WR1 1440
#define P_BR1 1472
#define P_WR2 1504
#define P_BR2 2016
#define P_WR3 2032
#define P_BR3 2048

// ===== fragment loaders (CDNA5 wave32 WMMA 16x16x32 f16 layouts) =====
// A (16x32, MxK): lanes 0-15 row M=lane hold K=[k0..k0+8) then [k0+16..k0+24);
// lanes 16-31 hold the +8 shifted runs.  Two 16B LDS loads per lane.
__device__ __forceinline__ v16h frag_a_lds(const _Float16* base, int strideH, int m0, int k0) {
  int lane = threadIdx.x & 31;
  const _Float16* p = base + (m0 + (lane & 15)) * strideH + k0 + ((lane >> 4) << 3);
  v8h lo = *(const v8h*)p;
  v8h hi = *(const v8h*)(p + 16);
  return __builtin_shufflevector(lo, hi, 0,1,2,3,4,5,6,7,8,9,10,11,12,13,14,15);
}
// B (32x16, KxN) with W stored [N][Kpad]: lane n<16 -> column n, K=[k0..k0+8) then [k0+16..k0+24);
// lanes 16-31 -> column n-16, the +8 shifted runs. Two 16B loads per lane.
__device__ __forceinline__ v16h frag_b_glb(const _Float16* __restrict__ W, int Kpad, int n0, int k0) {
  int lane = threadIdx.x & 31;
  const _Float16* p = W + (n0 + (lane & 15)) * Kpad + k0 + ((lane >> 4) << 3);
  v8h lo = *(const v8h*)p;
  v8h hi = *(const v8h*)(p + 16);
  return __builtin_shufflevector(lo, hi, 0,1,2,3,4,5,6,7,8,9,10,11,12,13,14,15);
}

// C/D 16x16 f32: VGPR r, lane l -> (m = m0 + (l>>4)*8 + r, n = n0 + (l&15))
__device__ __forceinline__ void store_act(_Float16* dst, int strideH, int m0, int n0,
                                          const v8f acc, const float* bias,
                                          const float* a, const float* c) {
  int lane = threadIdx.x & 31;
  int n = n0 + (lane & 15);
  int mb = m0 + ((lane >> 4) << 3);
  float bn = bias[n], an = a[n], cn = c[n];
#pragma unroll
  for (int r = 0; r < 8; ++r) {
    float h = acc[r] + bn;
    h = h > 0.f ? h : 0.f;
    h = an * h + cn;
    dst[(mb + r) * strideH + n] = (_Float16)h;
  }
}

#define WMMA_F16(A, B, C) \
  __builtin_amdgcn_wmma_f32_16x16x32_f16(false, (A), false, (B), (short)0, (C), false, false)

// ===================== prep: weights -> f16 [N][Kpad], BN folding =====================
__global__ void __launch_bounds__(256) ndt_prep(
    const float* __restrict__ W1, const float* __restrict__ W2,
    const float* __restrict__ W3, const float* __restrict__ Wn,
    const float* __restrict__ g1, const float* __restrict__ be1,
    const float* __restrict__ m1, const float* __restrict__ v1,
    const float* __restrict__ g2, const float* __restrict__ be2,
    const float* __restrict__ m2, const float* __restrict__ v2,
    const float* __restrict__ g3, const float* __restrict__ be3,
    const float* __restrict__ m3, const float* __restrict__ v3,
    char* __restrict__ ws) {
  int idx = blockIdx.x * 256 + threadIdx.x;
  _Float16* W1h = (_Float16*)(ws + WS_W1H);
  _Float16* W2h = (_Float16*)(ws + WS_W2H);
  _Float16* W3h = (_Float16*)(ws + WS_W3H);
  _Float16* Wnh = (_Float16*)(ws + WS_WNH);
  float* a1 = (float*)(ws + WS_A1); float* c1 = (float*)(ws + WS_C1);
  float* a2 = (float*)(ws + WS_A2); float* c2 = (float*)(ws + WS_C2);
  float* a3 = (float*)(ws + WS_A3); float* c3 = (float*)(ws + WS_C3);

  if (idx < 131072) {                      // W1h [256][512]
    int n = idx >> 9, k = idx & 511;
    W1h[idx] = (_Float16)((k < IN_DIM) ? W1[k * H1 + n] : 0.f);
  } else if (idx < 131072 + 32768) {       // W2h [128][256]
    int i = idx - 131072; int n = i >> 8, k = i & 255;
    W2h[i] = (_Float16)W2[k * H2 + n];
  } else if (idx < 163840 + 8192) {        // W3h [64][128]
    int i = idx - 163840; int n = i >> 7, k = i & 127;
    W3h[i] = (_Float16)W3[k * H3 + n];
  } else if (idx < 172032 + 2048) {        // Wnh [32][64] (only cols 0..4 used later)
    int i = idx - 172032; int n = i >> 6, k = i & 63;
    Wnh[i] = (_Float16)((n < 31) ? Wn[k * 31 + n] : 0.f);
  } else if (idx < 174080 + 256) {
    int i = idx - 174080;
    float a = g1[i] * rsqrtf(v1[i] + BN_EPS);
    a1[i] = a; c1[i] = be1[i] - a * m1[i];
  } else if (idx < 174336 + 128) {
    int i = idx - 174336;
    float a = g2[i] * rsqrtf(v2[i] + BN_EPS);
    a2[i] = a; c2[i] = be2[i] - a * m2[i];
  } else if (idx < 174464 + 64) {
    int i = idx - 174464;
    float a = g3[i] * rsqrtf(v3[i] + BN_EPS);
    a3[i] = a; c3[i] = be3[i] - a * m3[i];
  }
}

// ===================== fused MLP + soft tree + regressor =====================
__global__ void __launch_bounds__(256, 1) ndt_fused(
    const float* __restrict__ x,
    const float* __restrict__ b1, const float* __restrict__ b2, const float* __restrict__ b3,
    const float* __restrict__ thr, const float* __restrict__ temp, const float* __restrict__ leaf,
    const float* __restrict__ Wr1, const float* __restrict__ br1,
    const float* __restrict__ Wr2, const float* __restrict__ br2,
    const float* __restrict__ Wr3, const float* __restrict__ br3,
    const char* __restrict__ ws, float* __restrict__ out) {
  extern __shared__ char smem[];
  _Float16* act1 = (_Float16*)(smem + SM_A);      // stride 264 halves
  _Float16* bufB = (_Float16*)(smem + SM_B);      // stride 136 halves (xbuf, then act2)
  _Float16* act3 = (_Float16*)(smem + SM_A);      // stride 72 halves (reuses region A)
  float* nodes   = (float*)(smem + SM_NODES);     // 128 x 8 f32
  float* P       = (float*)(smem + SM_P);

  const _Float16* W1h = (const _Float16*)(ws + WS_W1H);
  const _Float16* W2h = (const _Float16*)(ws + WS_W2H);
  const _Float16* W3h = (const _Float16*)(ws + WS_W3H);
  const _Float16* Wnh = (const _Float16*)(ws + WS_WNH);
  const float* a1p = (const float*)(ws + WS_A1); const float* c1p = (const float*)(ws + WS_C1);
  const float* a2p = (const float*)(ws + WS_A2); const float* c2p = (const float*)(ws + WS_C2);
  const float* a3p = (const float*)(ws + WS_A3); const float* c3p = (const float*)(ws + WS_C3);

  const int tid  = threadIdx.x;
  const int wave = tid >> 5;
  const int lane = tid & 31;
  const int row0 = blockIdx.x * 128;

  // ---- preload scalar params into LDS ----
  {
    int i = tid;
    if (i < 256) { P[P_B1 + i] = b1[i]; P[P_A1 + i] = a1p[i]; P[P_C1 + i] = c1p[i]; }
    if (i < 128) { P[P_B2 + i] = b2[i]; P[P_A2 + i] = a2p[i]; P[P_C2 + i] = c2p[i]; }
    if (i < 64)  { P[P_B3 + i] = b3[i]; P[P_A3 + i] = a3p[i]; P[P_C3 + i] = c3p[i]; }
    if (i < 32) {
      P[P_THR + i]  = (i < 31) ? thr[i]  : 0.f;
      P[P_TEMP + i] = (i < 31) ? temp[i] : 1.f;
      P[P_LEAF + i] = leaf[i];
      P[P_WR1 + i]  = Wr1[i];
      P[P_BR1 + i]  = br1[i];
    }
    for (int j = i; j < 512; j += 256) P[P_WR2 + j] = Wr2[j];
    if (i < 16) { P[P_BR2 + i] = br2[i]; P[P_WR3 + i] = Wr3[i]; }
    if (i == 0) P[P_BR3] = br3[0];
  }

  // ================= GEMM1: [128 x 500] @ [500 x 256], K padded to 512 =================
  v8f acc1[8][2];
#pragma unroll
  for (int mt = 0; mt < 8; ++mt) { acc1[mt][0] = (v8f){}; acc1[mt][1] = (v8f){}; }

  for (int kc = 0; kc < 4; ++kc) {
    const int kbase = kc * 128;
    // stage + convert x chunk -> LDS f16 [128][136]
    // IN_DIM % 4 == 0 and x rows are 16B aligned (500*4 = 2000B stride), so the
    // tail guard collapses to a whole-float4 test: one aligned b128 load per 4 elems.
    for (int idx = tid; idx < 128 * 32; idx += 256) {
      int r  = idx >> 5;
      int kq = (idx & 31) << 2;
      int kg = kbase + kq;
      v4h hv = { (_Float16)0.f, (_Float16)0.f, (_Float16)0.f, (_Float16)0.f };
      if (kg < IN_DIM) {
        v4f f = *(const v4f*)(x + (size_t)(row0 + r) * IN_DIM + kg);
        hv = (v4h){ (_Float16)f.x, (_Float16)f.y, (_Float16)f.z, (_Float16)f.w };
      }
      *(v4h*)(bufB + r * 136 + kq) = hv;
    }
    __syncthreads();
#pragma unroll
    for (int kk = 0; kk < 4; ++kk) {
      const int k0 = kk * 32;
      v16h bf0 = frag_b_glb(W1h, K1PAD, (2 * wave + 0) * 16, kbase + k0);
      v16h bf1 = frag_b_glb(W1h, K1PAD, (2 * wave + 1) * 16, kbase + k0);
      // software-pipelined A fragments: a[mt+1] loads overlap wmma of a[mt]
      v16h a = frag_a_lds(bufB, 136, 0, k0);
#pragma unroll
      for (int mt = 0; mt < 8; ++mt) {
        v16h an = a;
        if (mt < 7) an = frag_a_lds(bufB, 136, (mt + 1) * 16, k0);
        acc1[mt][0] = WMMA_F16(a, bf0, acc1[mt][0]);
        acc1[mt][1] = WMMA_F16(a, bf1, acc1[mt][1]);
        a = an;
      }
    }
    __syncthreads();
  }
#pragma unroll
  for (int mt = 0; mt < 8; ++mt) {
    store_act(act1, 264, mt * 16, (2 * wave + 0) * 16, acc1[mt][0], P + P_B1, P + P_A1, P + P_C1);
    store_act(act1, 264, mt * 16, (2 * wave + 1) * 16, acc1[mt][1], P + P_B1, P + P_A1, P + P_C1);
  }
  __syncthreads();

  // ================= GEMM2: [128 x 256] @ [256 x 128] =================
  v8f acc2[8];
#pragma unroll
  for (int mt = 0; mt < 8; ++mt) acc2[mt] = (v8f){};
#pragma unroll
  for (int ks = 0; ks < 8; ++ks) {
    v16h b = frag_b_glb(W2h, 256, wave * 16, ks * 32);
    v16h a = frag_a_lds(act1, 264, 0, ks * 32);
#pragma unroll
    for (int mt = 0; mt < 8; ++mt) {
      v16h an = a;
      if (mt < 7) an = frag_a_lds(act1, 264, (mt + 1) * 16, ks * 32);
      acc2[mt] = WMMA_F16(a, b, acc2[mt]);
      a = an;
    }
  }
#pragma unroll
  for (int mt = 0; mt < 8; ++mt)
    store_act(bufB, 136, mt * 16, wave * 16, acc2[mt], P + P_B2, P + P_A2, P + P_C2);
  __syncthreads();

  // ================= GEMM3: [128 x 128] @ [128 x 64] =================
  v8f acc3[4];
#pragma unroll
  for (int mt = 0; mt < 4; ++mt) acc3[mt] = (v8f){};
  const int nt3 = wave & 3;
  const int mh3 = (wave >> 2) * 64;
#pragma unroll
  for (int ks = 0; ks < 4; ++ks) {
    v16h b = frag_b_glb(W3h, 128, nt3 * 16, ks * 32);
    v16h a = frag_a_lds(bufB, 136, mh3, ks * 32);
#pragma unroll
    for (int mt = 0; mt < 4; ++mt) {
      v16h an = a;
      if (mt < 3) an = frag_a_lds(bufB, 136, mh3 + (mt + 1) * 16, ks * 32);
      acc3[mt] = WMMA_F16(a, b, acc3[mt]);
      a = an;
    }
  }
#pragma unroll
  for (int mt = 0; mt < 4; ++mt)
    store_act(act3, 72, mh3 + mt * 16, nt3 * 16, acc3[mt], P + P_B3, P + P_A3, P + P_C3);
  __syncthreads();

  // ================= node GEMM: [128 x 64] @ [64 x 16] (only cols 0..4 used) =========
  {
    v8f acc4 = (v8f){};
#pragma unroll
    for (int ks = 0; ks < 2; ++ks) {
      v16h b = frag_b_glb(Wnh, 64, 0, ks * 32);
      v16h a = frag_a_lds(act3, 72, wave * 16, ks * 32);
      acc4 = WMMA_F16(a, b, acc4);
    }
    int n  = lane & 15;
    int mb = wave * 16 + ((lane >> 4) << 3);
    if (n < 5) {
#pragma unroll
      for (int r = 0; r < 8; ++r) nodes[(mb + r) * 8 + n] = acc4[r];
    }
  }
  __syncthreads();

  // ================= tail: sigmoid, 32-leaf soft tree, tiny MLP =================
  if (tid < 128) {
    float d[5];
#pragma unroll
    for (int i = 0; i < 5; ++i) {
      float z = (nodes[tid * 8 + i] - P[P_THR + i]) / P[P_TEMP + i];
      d[i] = 1.f / (1.f + expf(-z));
    }
    float p = 0.f;
#pragma unroll
    for (int l = 0; l < 32; ++l) {
      float pr = 1.f;
#pragma unroll
      for (int i = 0; i < 5; ++i) {
        float di = d[i];
        pr *= ((l >> (4 - i)) & 1) ? di : (1.f - di);
      }
      p += pr * P[P_LEAF + l];
    }
    float h1[32];
#pragma unroll
    for (int j = 0; j < 32; ++j) {
      float t = p * P[P_WR1 + j] + P[P_BR1 + j];
      h1[j] = t > 0.f ? t : 0.f;
    }
    float o = P[P_BR3];
#pragma unroll
    for (int k = 0; k < 16; ++k) {
      float s = P[P_BR2 + k];
#pragma unroll
      for (int j = 0; j < 32; ++j) s += h1[j] * P[P_WR2 + j * 16 + k];
      s = s > 0.f ? s : 0.f;
      o += s * P[P_WR3 + k];
    }
    out[row0 + tid] = o;
  }
}

extern "C" void kernel_launch(void* const* d_in, const int* in_sizes, int n_in,
                              void* d_out, int out_size, void* d_ws, size_t ws_size,
                              hipStream_t stream) {
  const float* x   = (const float*)d_in[0];
  const float* W1  = (const float*)d_in[1];
  const float* b1  = (const float*)d_in[2];
  const float* g1  = (const float*)d_in[3];
  const float* be1 = (const float*)d_in[4];
  const float* m1  = (const float*)d_in[5];
  const float* v1  = (const float*)d_in[6];
  const float* W2  = (const float*)d_in[7];
  const float* b2  = (const float*)d_in[8];
  const float* g2  = (const float*)d_in[9];
  const float* be2 = (const float*)d_in[10];
  const float* m2  = (const float*)d_in[11];
  const float* v2  = (const float*)d_in[12];
  const float* W3  = (const float*)d_in[13];
  const float* b3  = (const float*)d_in[14];
  const float* g3  = (const float*)d_in[15];
  const float* be3 = (const float*)d_in[16];
  const float* m3  = (const float*)d_in[17];
  const float* v3  = (const float*)d_in[18];
  const float* Wn  = (const float*)d_in[19];
  const float* thr = (const float*)d_in[20];
  const float* tmp = (const float*)d_in[21];
  const float* lv  = (const float*)d_in[22];
  const float* Wr1 = (const float*)d_in[23];
  const float* br1 = (const float*)d_in[24];
  const float* Wr2 = (const float*)d_in[25];
  const float* br2 = (const float*)d_in[26];
  const float* Wr3 = (const float*)d_in[27];
  const float* br3 = (const float*)d_in[28];
  char* ws = (char*)d_ws;
  float* out = (float*)d_out;

  ndt_prep<<<682, 256, 0, stream>>>(W1, W2, W3, Wn,
                                    g1, be1, m1, v1,
                                    g2, be2, m2, v2,
                                    g3, be3, m3, v3, ws);
  ndt_fused<<<512, 256, SMEM_BYTES, stream>>>(x, b1, b2, b3, thr, tmp, lv,
                                              Wr1, br1, Wr2, br2, Wr3, br3, ws, out);
}
